// SimpleNet_65798898975281
// MI455X (gfx1250) — matrix operand order repaired and checked by reading, not compiled, not run on hardware
//
#include <hip/hip_runtime.h>
#include <hip/hip_bf16.h>
#include <math.h>

typedef __bf16 bf16_t;
typedef bf16_t v16bf __attribute__((ext_vector_type(16)));
typedef float  v8f   __attribute__((ext_vector_type(8)));

#define CIN   256
#define COUT  512
#define HDIM  64
#define WDIM  64
#define HW    4096          // 64*64
#define NB    4
#define NTOT  (NB * HW)     // 16384 pixels total (GEMM N)
#define NSP   64500
#define NFA   510
#define NOR   100
#define NROW  (NSP + NFA + NOR)   // 65110

// ---------------------------------------------------------------------------
// Kernel 1: fused Q/K/V 1x1-conv GEMM with bf16 WMMA, f32 accumulate.
//  C[o, n] = sum_c W[o, c] * X[c, n],  n = b*4096 + p
//  One wave computes one 16x16 output tile for q, k and v simultaneously
//  (B-fragment of x is loaded once, reused for three WMMAs).
// ---------------------------------------------------------------------------
__device__ __forceinline__ void load_afrag(const float* __restrict__ wr, v16bf& a)
{
    // wr is 32-byte aligned: row*1024B + kk*4B (128B) + hi*32B.
    const float4 f0 = *reinterpret_cast<const float4*>(wr);        // K = base+0..3
    const float4 f1 = *reinterpret_cast<const float4*>(wr + 4);    // K = base+4..7
    const float4 f2 = *reinterpret_cast<const float4*>(wr + 16);   // K = base+16..19
    const float4 f3 = *reinterpret_cast<const float4*>(wr + 20);   // K = base+20..23
    a[0]  = (bf16_t)f0.x;  a[1]  = (bf16_t)f0.y;  a[2]  = (bf16_t)f0.z;  a[3]  = (bf16_t)f0.w;
    a[4]  = (bf16_t)f1.x;  a[5]  = (bf16_t)f1.y;  a[6]  = (bf16_t)f1.z;  a[7]  = (bf16_t)f1.w;
    a[8]  = (bf16_t)f2.x;  a[9]  = (bf16_t)f2.y;  a[10] = (bf16_t)f2.z;  a[11] = (bf16_t)f2.w;
    a[12] = (bf16_t)f3.x;  a[13] = (bf16_t)f3.y;  a[14] = (bf16_t)f3.z;  a[15] = (bf16_t)f3.w;
}

__global__ __launch_bounds__(128) void qkv_wmma_kernel(
    const float* __restrict__ x,
    const float* __restrict__ wq, const float* __restrict__ bq,
    const float* __restrict__ wk, const float* __restrict__ bk,
    const float* __restrict__ wv, const float* __restrict__ bv,
    float* __restrict__ qo, float* __restrict__ ko, float* __restrict__ vo)
{
    const int lane  = threadIdx.x & 31;
    const int wid   = threadIdx.x >> 5;
    const int col   = lane & 15;      // N column / A row within tile
    const int hi    = lane >> 4;      // half-wave select

    const int n0 = (blockIdx.x * 4 + wid) * 16;   // pixel tile (never crosses b: 4096 % 16 == 0)
    const int m0 = blockIdx.y * 16;               // out-channel tile
    const int b  = n0 >> 12;
    const int p0 = n0 & (HW - 1);

    const float* __restrict__ xb = x + (size_t)b * CIN * HW;

    v8f accq = {}; v8f acck = {}; v8f accv = {};

    const int arow = m0 + col;        // weight row owned by this lane

    for (int kk = 0; kk < CIN; kk += 32) {
        // ---- B fragment (32x16 slice of X), bf16.  Lane = column N=p0+col;
        //      elems 0..7 -> K = kk + hi*8 + e, elems 8..15 -> K = kk + hi*8 + 16 + (e-8).
        v16bf bfrag;
        #pragma unroll
        for (int e = 0; e < 16; ++e) {
            const int kg = kk + hi * 8 + (e & 7) + (e >> 3) * 16;
            bfrag[e] = (bf16_t)xb[(size_t)kg * HW + p0 + col];
        }

        // ---- A fragments (16x32 slices of Wq/Wk/Wv), bf16, via b128 loads.
        //      Lane owns row arow; elems 0..7 -> K = kk + hi*8 + e,
        //      elems 8..15 -> K = kk + hi*8 + 16 + (e-8).
        v16bf aq, ak, av;
        load_afrag(wq + (size_t)arow * CIN + kk + hi * 8, aq);
        load_afrag(wk + (size_t)arow * CIN + kk + hi * 8, ak);
        load_afrag(wv + (size_t)arow * CIN + kk + hi * 8, av);

        accq = __builtin_amdgcn_wmma_f32_16x16x32_bf16(false, aq, false, bfrag,
                                                       (short)0, accq, false, false);
        acck = __builtin_amdgcn_wmma_f32_16x16x32_bf16(false, ak, false, bfrag,
                                                       (short)0, acck, false, false);
        accv = __builtin_amdgcn_wmma_f32_16x16x32_bf16(false, av, false, bfrag,
                                                       (short)0, accv, false, false);
    }

    // ---- Store: VGPR r holds (M = m0 + r + 8*hi, N = p0 + col)
    #pragma unroll
    for (int r = 0; r < 8; ++r) {
        const int    o   = m0 + r + hi * 8;
        const size_t idx = (size_t)b * COUT * HW + (size_t)o * HW + p0 + col;
        qo[idx] = accq[r] + bq[o];
        ko[idx] = acck[r] + bk[o];
        vo[idx] = accv[r] + bv[o];
    }
}

// ---------------------------------------------------------------------------
// Kernel 2: per-channel 3x3 window softmax attention + ReLU + mean pool.
//  One workgroup per (b, c) plane; k/v planes staged in LDS (2 x 16 KB) via
//  CDNA5 async memory->LDS copies (global_load_async_to_lds_b128, ASYNCcnt),
//  bypassing VGPRs.  Zero-padding semantics: k padded 0 BEFORE rel is added.
// ---------------------------------------------------------------------------
__global__ __launch_bounds__(256) void attn_pool_kernel(
    const float* __restrict__ q, const float* __restrict__ k,
    const float* __restrict__ v,
    const float* __restrict__ rel_h, const float* __restrict__ rel_w,
    float* __restrict__ pooled)
{
    __shared__ float sk[HW];
    __shared__ float sv[HW];
    __shared__ float red[256];

    const int bc = blockIdx.x;           // b * COUT + c
    const int c  = bc & (COUT - 1);
    const size_t base = (size_t)bc * HW;

    // Async copy of the two 16 KB planes into LDS.  Each thread moves 16 B per
    // issue; 4 issues per plane.  LDS byte address = low 32 bits of the flat
    // pointer (LDS aperture mapping), global base is block-uniform -> SGPR.
    {
        const float* __restrict__ ksrc = k + base;
        const float* __restrict__ vsrc = v + base;
        #pragma unroll
        for (int it = 0; it < 4; ++it) {
            const int      fi    = (it * 256 + (int)threadIdx.x) * 4;  // float index
            const unsigned goff  = (unsigned)fi * 4u;                  // byte offset
            const unsigned lds_k = (unsigned)(uintptr_t)&sk[fi];
            const unsigned lds_v = (unsigned)(uintptr_t)&sv[fi];
            asm volatile("global_load_async_to_lds_b128 %0, %1, %2"
                         :: "v"(lds_k), "v"(goff), "s"(ksrc) : "memory");
            asm volatile("global_load_async_to_lds_b128 %0, %1, %2"
                         :: "v"(lds_v), "v"(goff), "s"(vsrc) : "memory");
        }
        asm volatile("s_wait_asynccnt 0x0" ::: "memory");
    }
    __syncthreads();

    // rel term per window position (uniform across the block)
    float rel[3][3];
    if (c < COUT / 2) {
        #pragma unroll
        for (int i = 0; i < 3; ++i) {
            const float rh = rel_h[c * 3 + i];
            #pragma unroll
            for (int j = 0; j < 3; ++j) rel[i][j] = rh;
        }
    } else {
        #pragma unroll
        for (int j = 0; j < 3; ++j) {
            const float rw = rel_w[(c - COUT / 2) * 3 + j];
            #pragma unroll
            for (int i = 0; i < 3; ++i) rel[i][j] = rw;
        }
    }

    float acc = 0.0f;
    for (int pix = threadIdx.x; pix < HW; pix += 256) {
        const int h = pix >> 6;
        const int w = pix & 63;
        const float qv = q[base + pix];

        float lg[9], vv[9];
        float mx = -INFINITY;
        #pragma unroll
        for (int i = 0; i < 3; ++i) {
            #pragma unroll
            for (int j = 0; j < 3; ++j) {
                const int hh = h + i - 1;
                const int ww = w + j - 1;
                const bool in = ((unsigned)hh < (unsigned)HDIM) &&
                                ((unsigned)ww < (unsigned)WDIM);
                const float kv = in ? sk[hh * WDIM + ww] : 0.0f;
                vv[i * 3 + j]  = in ? sv[hh * WDIM + ww] : 0.0f;
                const float l  = qv * (kv + rel[i][j]);
                lg[i * 3 + j]  = l;
                mx = fmaxf(mx, l);
            }
        }
        float s = 0.0f, o = 0.0f;
        #pragma unroll
        for (int t = 0; t < 9; ++t) {
            const float e = __expf(lg[t] - mx);
            s += e;
            o += e * vv[t];
        }
        acc += fmaxf(o / s, 0.0f);     // ReLU then accumulate for mean
    }

    red[threadIdx.x] = acc;
    __syncthreads();
    for (int st = 128; st > 0; st >>= 1) {
        if (threadIdx.x < st) red[threadIdx.x] += red[threadIdx.x + st];
        __syncthreads();
    }
    if (threadIdx.x == 0) pooled[bc] = red[0] * (1.0f / (float)HW);
}

// ---------------------------------------------------------------------------
// Kernel 3: three classification heads as a 4-row GEMV over 65110x512 weights.
//  Bandwidth-bound (133 MB weights): one wave per output row, wave32 shuffle
//  reduction, pooled (8 KB) staged in LDS.
// ---------------------------------------------------------------------------
__global__ __launch_bounds__(256) void heads_kernel(
    const float* __restrict__ pooled,
    const float* __restrict__ wsp, const float* __restrict__ bsp,
    const float* __restrict__ wfa, const float* __restrict__ bfa,
    const float* __restrict__ wod, const float* __restrict__ bod,
    float* __restrict__ out)
{
    __shared__ float sp[NB * COUT];
    for (int i = threadIdx.x; i < NB * COUT; i += 256) sp[i] = pooled[i];
    __syncthreads();

    const int lane = threadIdx.x & 31;
    const int wid  = threadIdx.x >> 5;
    const int j    = blockIdx.x * 8 + wid;
    if (j >= NROW) return;

    const float* __restrict__ wrow;
    float bias;
    if (j < NSP) {
        wrow = wsp + (size_t)j * COUT;            bias = bsp[j];
    } else if (j < NSP + NFA) {
        const int t = j - NSP;
        wrow = wfa + (size_t)t * COUT;            bias = bfa[t];
    } else {
        const int t = j - NSP - NFA;
        wrow = wod + (size_t)t * COUT;            bias = bod[t];
    }

    float a0 = 0.f, a1 = 0.f, a2 = 0.f, a3 = 0.f;
    for (int ci = lane; ci < COUT; ci += 32) {
        __builtin_prefetch(&wrow[ci + 128], 0, 1);   // global_prefetch_b8
        const float wv_ = wrow[ci];
        a0 += wv_ * sp[ci];
        a1 += wv_ * sp[COUT + ci];
        a2 += wv_ * sp[2 * COUT + ci];
        a3 += wv_ * sp[3 * COUT + ci];
    }
    #pragma unroll
    for (int off = 16; off > 0; off >>= 1) {
        a0 += __shfl_xor(a0, off);
        a1 += __shfl_xor(a1, off);
        a2 += __shfl_xor(a2, off);
        a3 += __shfl_xor(a3, off);
    }
    if (lane == 0) {
        out[(size_t)0 * NROW + j] = a0 + bias;
        out[(size_t)1 * NROW + j] = a1 + bias;
        out[(size_t)2 * NROW + j] = a2 + bias;
        out[(size_t)3 * NROW + j] = a3 + bias;
    }
}

// ---------------------------------------------------------------------------
extern "C" void kernel_launch(void* const* d_in, const int* in_sizes, int n_in,
                              void* d_out, int out_size, void* d_ws, size_t ws_size,
                              hipStream_t stream)
{
    const float* x    = (const float*)d_in[0];
    const float* wq   = (const float*)d_in[1];
    const float* bq   = (const float*)d_in[2];
    const float* wk   = (const float*)d_in[3];
    const float* bk   = (const float*)d_in[4];
    const float* wv   = (const float*)d_in[5];
    const float* bv   = (const float*)d_in[6];
    const float* relh = (const float*)d_in[7];
    const float* relw = (const float*)d_in[8];
    const float* wsp  = (const float*)d_in[9];
    const float* bsp  = (const float*)d_in[10];
    const float* wfa  = (const float*)d_in[11];
    const float* bfa  = (const float*)d_in[12];
    const float* wod  = (const float*)d_in[13];
    const float* bod  = (const float*)d_in[14];

    const size_t plane_elems = (size_t)NB * COUT * HW;   // 8,388,608 floats each
    float* qb     = (float*)d_ws;
    float* kb     = qb + plane_elems;
    float* vb     = kb + plane_elems;
    float* pooled = vb + plane_elems;                    // 2048 floats

    // Stage 1: QKV WMMA GEMM. 1024 N-tiles x 32 M-tiles, 4 waves/block.
    dim3 g1(NTOT / 16 / 4, COUT / 16);                   // (256, 32)
    qkv_wmma_kernel<<<g1, 128, 0, stream>>>(x, wq, bq, wk, bk, wv, bv, qb, kb, vb);

    // Stage 2: windowed softmax attention + ReLU + mean pool; 1 block per (b,c).
    attn_pool_kernel<<<NB * COUT, 256, 0, stream>>>(qb, kb, vb, relh, relw, pooled);

    // Stage 3: heads GEMV; 8 rows (waves) per block.
    heads_kernel<<<(NROW + 7) / 8, 256, 0, stream>>>(pooled, wsp, bsp, wfa, bfa,
                                                     wod, bod, (float*)d_out);
}
